// Dcls1d_82927228552100
// MI455X (gfx1250) — matrix operand order repaired and checked by reading, not compiled, not run on hardware
//
#include <hip/hip_runtime.h>
#include <hip/hip_bf16.h>

// ---------------- problem constants ----------------
#define OUT_C 256
#define IN_C  256
#define KC    26
#define KD    25
#define BATCH 32
#define LEN   1024
#define LOUT  (LEN - KD + 1)   // 1000
#define KTOT  (IN_C * KD)      // 6400
#define NCHUNK (KTOT / 32)     // 200 chunks of K=32

typedef __attribute__((ext_vector_type(16))) __bf16 v16bf;
typedef __attribute__((ext_vector_type(8)))  __bf16 v8bf;
typedef __attribute__((ext_vector_type(8)))  float  v8f;
typedef __attribute__((ext_vector_type(4)))  int    v4i;

// ---------------- CDNA5 async global->LDS (probe-confirmed present) ----
#if defined(__has_builtin)
#if __has_builtin(__builtin_amdgcn_global_load_async_to_lds_b128)
#define HAVE_ASYNC_LDS 1
#endif
#endif

#ifdef HAVE_ASYNC_LDS
typedef __attribute__((address_space(1))) v4i gv4i;   // global int4
typedef __attribute__((address_space(3))) v4i lv4i;   // LDS int4
#endif

__device__ __forceinline__ void copy16_to_lds(const void* g, void* l) {
#ifdef HAVE_ASYNC_LDS
    __builtin_amdgcn_global_load_async_to_lds_b128(
        (gv4i*)(uintptr_t)g,
        (lv4i*)(unsigned)(uintptr_t)l,
        0, 0);
#else
    *(v8bf*)l = *(const v8bf*)g;
#endif
}

__device__ __forceinline__ void async_stage_wait() {
#ifdef HAVE_ASYNC_LDS
#if __has_builtin(__builtin_amdgcn_s_wait_asynccnt)
    __builtin_amdgcn_s_wait_asynccnt(0);
#else
    asm volatile("s_wait_asynccnt 0x0" ::: "memory");
#endif
#endif
}

// =====================================================================
// Kernel 1: Gaussian kernel construction (fp32 math, bf16 output)
// Kbf layout: [d][o][i]  (so GEMM A-tile rows are contiguous in i)
// =====================================================================
__global__ __launch_bounds__(IN_C)
void build_kernel_gauss(const float* __restrict__ W,
                        const float* __restrict__ sgn,
                        const float* __restrict__ P,
                        const float* __restrict__ SIG,
                        __bf16* __restrict__ Kbf) {
    const int o = blockIdx.x;
    const int i = threadIdx.x;
    const size_t base = ((size_t)o * IN_C + i) * KC;

    float acc[KD];
#pragma unroll
    for (int d = 0; d < KD; ++d) acc[d] = 0.0f;

    for (int c = 0; c < KC; ++c) {
        const float w  = W[base + c] * sgn[base + c];
        const float pc = P[base + c] + (float)(KD / 2);
        const float s  = fabsf(SIG[base + c]) + 0.27f;
        const float inv = 1.0f / s;
        float X[KD];
        float sum = 0.0f;
#pragma unroll
        for (int d = 0; d < KD; ++d) {
            const float t = ((float)d - pc) * inv;
            const float e = __expf(-0.5f * t * t);
            X[d] = e;
            sum += e;
        }
        const float r = w / (sum + 1e-7f);
#pragma unroll
        for (int d = 0; d < KD; ++d) acc[d] += X[d] * r;
    }
#pragma unroll
    for (int d = 0; d < KD; ++d)
        Kbf[((size_t)d * OUT_C + o) * IN_C + i] = (__bf16)acc[d];
}

// =====================================================================
// Kernel 2: transpose x (B,I,L) f32 -> xT (B,L,I) bf16 (coalesced both sides)
// =====================================================================
__global__ __launch_bounds__(256)
void transpose_x_bf16(const float* __restrict__ x, __bf16* __restrict__ xT) {
    __shared__ float tile[32][33];
    const int b  = blockIdx.z;
    const int i0 = blockIdx.y * 32;
    const int l0 = blockIdx.x * 32;
    const int tx = threadIdx.x;      // 0..31
    const int ty = threadIdx.y;      // 0..7
#pragma unroll
    for (int r = 0; r < 4; ++r) {
        const int i = i0 + ty + r * 8;
        tile[ty + r * 8][tx] = x[((size_t)b * IN_C + i) * LEN + l0 + tx];
    }
    __syncthreads();
#pragma unroll
    for (int r = 0; r < 4; ++r) {
        const int l = l0 + ty + r * 8;
        xT[((size_t)b * LEN + l) * IN_C + i0 + tx] = (__bf16)tile[tx][ty + r * 8];
    }
}

// =====================================================================
// Kernel 3: implicit-GEMM conv using v_wmma_f32_16x16x32_bf16
//   out[b, o, l] = sum_{i,d} K[o,i,d] * x[b,i,l+d]
//   Block tile 128(M=o) x 128(N=l); 8 waves = 2(Mhalf) x 4(Nquarter)
//   Wave tile 64x32 -> 8 accumulators; A frag reused across 2 B frags.
//   Double-buffered LDS staged via async global->LDS (ASYNCcnt).
// =====================================================================
#define LDS_STRIDE 40   // 32 bf16 + pad (80B row -> spread banks for frag loads)

__global__ __launch_bounds__(256)
void dcls_conv_wmma(const __bf16* __restrict__ Kbf,
                    const __bf16* __restrict__ xT,
                    float* __restrict__ out) {
    __shared__ __align__(16) __bf16 lsA[2][128][LDS_STRIDE];
    __shared__ __align__(16) __bf16 lsB[2][128][LDS_STRIDE];

    const int tid  = threadIdx.x;
    const int lane = tid & 31;
    const int wave = tid >> 5;
    const int wm   = wave >> 2;          // 0..1 -> M half (64 rows of o)
    const int wn   = wave & 3;           // 0..3 -> N quarter (32 cols of l)
    const int o0   = blockIdx.x * 128;
    const int l0   = blockIdx.y * 128;
    const int b    = blockIdx.z;

    v8f zero = {0.f, 0.f, 0.f, 0.f, 0.f, 0.f, 0.f, 0.f};
    v8f acc[4][2];
#pragma unroll
    for (int s = 0; s < 4; ++s) { acc[s][0] = zero; acc[s][1] = zero; }

    // stage chunk kk into buffer buf:  d = kk>>3, i0k = (kk&7)*32
    // A: 128x32 bf16, B: 128x32 bf16; each thread moves 2x16B of each.
    auto stage = [&](int kk, int buf) {
        const int d   = kk >> 3;
        const int i0k = (kk & 7) << 5;
        const int r   = tid >> 1;
        const int off = (tid & 1) * 16;
        {
            const __bf16* g = Kbf + ((size_t)d * OUT_C + o0 + r) * IN_C + i0k + off;
            copy16_to_lds(g,     &lsA[buf][r][off]);
            copy16_to_lds(g + 8, &lsA[buf][r][off + 8]);
        }
        {
            int lr = l0 + d + r;
            if (lr > LEN - 1) lr = LEN - 1;     // clamp (stores are masked)
            const __bf16* g = xT + ((size_t)b * LEN + lr) * IN_C + i0k + off;
            copy16_to_lds(g,     &lsB[buf][r][off]);
            copy16_to_lds(g + 8, &lsB[buf][r][off + 8]);
        }
    };

    // per-lane fragment addressing (contiguous K = 16*(lane/16) + s, same in A and B, wave32)
    const int mrow = lane & 15;
    const int klo  = (lane < 16) ? 0 : 16;    // A: 16 consecutive K at klo
    const int kb   = (lane < 16) ? 0 : 16;    // B: 16 consecutive K at kb
    const int nl0  = wn * 32 + (lane & 15);   // B columns: n-subtile 0
    const int nl1  = nl0 + 16;                //             n-subtile 1

    union Frag { v16bf v; v8bf h[2]; };

    stage(0, 0);
    async_stage_wait();
    __syncthreads();

    for (int kk = 0; kk < NCHUNK; ++kk) {
        const int buf = kk & 1;
        if (kk + 1 < NCHUNK) stage(kk + 1, buf ^ 1);

        Frag fb0, fb1;
        fb0.h[0] = *(const v8bf*)&lsB[buf][nl0][kb];
        fb0.h[1] = *(const v8bf*)&lsB[buf][nl0][kb + 8];
        fb1.h[0] = *(const v8bf*)&lsB[buf][nl1][kb];
        fb1.h[1] = *(const v8bf*)&lsB[buf][nl1][kb + 8];
#pragma unroll
        for (int s = 0; s < 4; ++s) {
            const int m = wm * 64 + s * 16 + mrow;
            Frag fa;
            fa.h[0] = *(const v8bf*)&lsA[buf][m][klo];
            fa.h[1] = *(const v8bf*)&lsA[buf][m][klo + 8];
            acc[s][0] = __builtin_amdgcn_wmma_f32_16x16x32_bf16(
                false, fa.v, false, fb0.v, (short)0, acc[s][0], false, false);
            acc[s][1] = __builtin_amdgcn_wmma_f32_16x16x32_bf16(
                false, fa.v, false, fb1.v, (short)0, acc[s][1], false, false);
        }

        async_stage_wait();   // our async stores to LDS done before barrier
        __syncthreads();      // all waves' staging of kk+1 visible; buf free
    }

    // C/D layout: lane -> N = lane&15; VGPR v -> M = v + 8*(lane>>4)
    const int mbase = o0 + wm * 64 + ((lane >> 4) << 3);
#pragma unroll
    for (int ns = 0; ns < 2; ++ns) {
        const int nout = l0 + wn * 32 + ns * 16 + (lane & 15);
        if (nout < LOUT) {
#pragma unroll
            for (int s = 0; s < 4; ++s) {
#pragma unroll
                for (int v = 0; v < 8; ++v) {
                    const int o = mbase + s * 16 + v;
                    out[((size_t)b * OUT_C + o) * LOUT + nout] = acc[s][ns][v];
                }
            }
        }
    }
}

// =====================================================================
// Host launcher
// =====================================================================
extern "C" void kernel_launch(void* const* d_in, const int* in_sizes, int n_in,
                              void* d_out, int out_size, void* d_ws, size_t ws_size,
                              hipStream_t stream) {
    const float* x   = (const float*)d_in[0];   // (32,256,1024)
    const float* W   = (const float*)d_in[1];   // (256,256,26)
    const float* sgn = (const float*)d_in[2];   // (256,256,26)
    const float* P   = (const float*)d_in[3];   // (1,256,256,26)
    const float* SIG = (const float*)d_in[4];   // (1,256,256,26)
    float* out = (float*)d_out;                 // (32,256,1000)

    __bf16* Kbf = (__bf16*)d_ws;                              // 25*256*256*2  = 3.28 MB
    __bf16* xT  = (__bf16*)((char*)d_ws + (size_t)(4 << 20)); // 32*1024*256*2 = 16.8 MB

    build_kernel_gauss<<<dim3(OUT_C), dim3(IN_C), 0, stream>>>(W, sgn, P, SIG, Kbf);
    transpose_x_bf16<<<dim3(LEN / 32, IN_C / 32, BATCH), dim3(32, 8), 0, stream>>>(x, xT);
    dcls_conv_wmma<<<dim3(OUT_C / 128, LEN / 128, BATCH), dim3(256), 0, stream>>>(Kbf, xT, out);
}
